// EfficientSelfAttention_64527588655263
// MI455X (gfx1250) — matrix operand order repaired and checked by reading, not compiled, not run on hardware
//
#include <hip/hip_runtime.h>
#include <hip/hip_bf16.h>

typedef __attribute__((ext_vector_type(16))) _Float16 v16h;
typedef __attribute__((ext_vector_type(8)))  float    v8f;

// -------------------------------------------------------------------------
// Async global->LDS stage of a 32 KB-per-call stride using CDNA5
// GLOBAL_LOAD_ASYNC_TO_LDS_B128 (ASYNCcnt-tracked, bypasses VGPRs).
// Each of the 256 threads issues 8 x 16B transfers (32 KB total).
// LDS operand = low 32 bits of the flat shared pointer (LDS byte offset).
// -------------------------------------------------------------------------
__device__ __forceinline__ void stage_lds_async_32k(_Float16* ldsDst,
                                                    const _Float16* gsrc,
                                                    int tid) {
    unsigned int ldsOff = (unsigned int)(uintptr_t)ldsDst + (unsigned int)tid * 16u;
    const char* g = (const char*)gsrc + (size_t)tid * 16u;
    #pragma unroll
    for (int i = 0; i < 8; ++i) {
        asm volatile("global_load_async_to_lds_b128 %0, %1, off"
                     :: "v"(ldsOff + (unsigned int)i * 4096u),
                        "v"(g + (size_t)i * 4096)
                     : "memory");
    }
    asm volatile("s_wait_asynccnt 0" ::: "memory");
}

// -------------------------------------------------------------------------
// Generic WMMA GEMM:  Y[M,N] = W[M,K] * X[K,N]   (f32 in, f16 WMMA, f32 acc)
// Block tile 128x128, 256 threads = 8 waves arranged 2(M) x 4(N),
// each wave computes 64x32 = 4x2 fragments of 16x16.
// LDS staging uses fragment-native layouts so each operand load is one
// contiguous 32B read per lane. (Staging keeps the VGPR path: it does
// f32->f16 conversion + layout swizzle which async DMA cannot.)
//
// BMODE: 0 = x / plain [b][K][4096] f32 source (N = b*4096 + hw)
//        1 = strided 2x2 conv gather from x (K = c*4+dy*2+dx, N = b*1024+oy*32+ox)
//        2 = same addressing as 0 (used for the fused img+batch sum buffer)
// OMODE: 0 = plain f32 [o][Ntot]
//        1 = qA f16 A-fragment layout  [bh][rowblk][lane][h]
//        2 = kv split: o<256 -> kB f16 B-frag layout (k-dim=channel),
//                      o>=256 -> vB f16 B-frag layout (k-dim=key)
// -------------------------------------------------------------------------
template<int BMODE, int OMODE>
__global__ __launch_bounds__(256) void gemm_wmma_k(
    const float* __restrict__ W, const float* __restrict__ X,
    void* __restrict__ out0, void* __restrict__ out1, int Kdim, int Ntot)
{
    __shared__ __align__(32) _Float16 As[8 * 32 * 16];  // 128x32 A tile, frag layout
    __shared__ __align__(32) _Float16 Bs[8 * 32 * 16];  // 32x128 B tile, frag layout

    const int tid  = threadIdx.x;
    const int lane = tid & 31;
    const int w    = tid >> 5;
    const int wm   = w >> 2;      // 0..1
    const int wn   = w & 3;       // 0..3
    const int oBlk = blockIdx.y * 128;
    const int nBlk = blockIdx.x * 128;

    v8f zero = {};
    v8f acc[4][2];
    #pragma unroll
    for (int i = 0; i < 4; ++i)
        #pragma unroll
        for (int j = 0; j < 2; ++j) acc[i][j] = zero;

    for (int k0 = 0; k0 < Kdim; k0 += 32) {
        __syncthreads();
        // ---- stage A (weights) into A-fragment layout ----
        #pragma unroll
        for (int r = 0; r < 16; ++r) {
            int idx = tid + r * 256;          // 0..4095
            int i = idx >> 5, j = idx & 31;   // row i (0..127), k j (0..31)
            float wv = W[(size_t)(oBlk + i) * Kdim + (k0 + j)];
            int rm = i >> 4, m = i & 15;
            int lp = m + (((j & 15) >= 8) ? 16 : 0);
            int h  = ((j >= 16) ? 8 : 0) + (j & 7);
            As[(rm * 32 + lp) * 16 + h] = (_Float16)wv;
        }
        // ---- stage B (activations) into B-fragment layout ----
        #pragma unroll
        for (int r = 0; r < 16; ++r) {
            int idx = tid + r * 256;
            int k = idx >> 7, n = idx & 127;  // k (0..31), n (0..127)
            float xv;
            if (BMODE == 1) {
                int ng = nBlk + n;
                int b = ng >> 10, rem = ng & 1023;
                int oy = rem >> 5, ox = rem & 31;
                int kg = k0 + k;
                int c = kg >> 2, dy = (kg >> 1) & 1, dx = kg & 1;
                xv = X[((size_t)(b * 256 + c) * 64 + (oy * 2 + dy)) * 64 + (ox * 2 + dx)];
            } else {
                int ng = nBlk + n;
                int b = ng >> 12, hw = ng & 4095;
                xv = X[(size_t)(b * 256 + (k0 + k)) * 4096 + hw];
            }
            int nt = n >> 4;
            int lp = (n & 15) + (((k & 15) >= 8) ? 16 : 0);
            int h  = ((k >= 16) ? 8 : 0) + (k & 7);
            Bs[(nt * 32 + lp) * 16 + h] = (_Float16)xv;
        }
        __syncthreads();

        v16h bfr[2];
        #pragma unroll
        for (int nf = 0; nf < 2; ++nf) {
            int nt = wn * 2 + nf;
            bfr[nf] = *(const v16h*)(&Bs[(nt * 32 + lane) * 16]);
        }
        #pragma unroll
        for (int mf = 0; mf < 4; ++mf) {
            int rm = wm * 4 + mf;
            v16h afr = *(const v16h*)(&As[(rm * 32 + lane) * 16]);
            #pragma unroll
            for (int nf = 0; nf < 2; ++nf) {
                acc[mf][nf] = __builtin_amdgcn_wmma_f32_16x16x32_f16(
                    false, afr, false, bfr[nf], (short)0, acc[mf][nf], false, false);
            }
        }
    }

    // ---- store C fragments ----
    #pragma unroll
    for (int mf = 0; mf < 4; ++mf) {
        #pragma unroll
        for (int nf = 0; nf < 2; ++nf) {
            int oBase = oBlk + wm * 64 + mf * 16;
            int nBase = nBlk + wn * 32 + nf * 16;
            #pragma unroll
            for (int r = 0; r < 8; ++r) {
                int m  = r + ((lane >= 16) ? 8 : 0);
                int nl = lane & 15;
                int o  = oBase + m;
                int ng = nBase + nl;
                float val = acc[mf][nf][r];
                if (OMODE == 0) {
                    ((float*)out0)[(size_t)o * Ntot + ng] = val;
                } else if (OMODE == 1) {
                    // qA: A-fragment layout keyed by (pos, channel-within-head)
                    int b = ng >> 12, pos = ng & 4095;
                    int head = o >> 5, c = o & 31;
                    int bh = b * 8 + head;
                    int rowblk = pos >> 4, mm = pos & 15;
                    int lp = mm + (((c & 15) >= 8) ? 16 : 0);
                    int h  = ((c >= 16) ? 8 : 0) + (c & 7);
                    ((_Float16*)out0)[((size_t)(bh * 256 + rowblk) * 32 + lp) * 16 + h] = (_Float16)val;
                } else {
                    int b = ng >> 10, key = ng & 1023;
                    if (o < 256) {
                        // K: B-fragment layout, GEMM k-dim = channel, n = key
                        int head = o >> 5, c = o & 31;
                        int bh = b * 8 + head;
                        int kt = key >> 4;
                        int lp = (key & 15) + (((c & 15) >= 8) ? 16 : 0);
                        int h  = ((c >= 16) ? 8 : 0) + (c & 7);
                        ((_Float16*)out0)[((size_t)(bh * 64 + kt) * 32 + lp) * 16 + h] = (_Float16)val;
                    } else {
                        // V: B-fragment layout, GEMM k-dim = key, n = channel
                        int o2 = o - 256;
                        int head = o2 >> 5, c = o2 & 31;
                        int bh = b * 8 + head;
                        int keyblk = key >> 5, kk = key & 31;
                        int ctile = c >> 4, col = c & 15;
                        int lp = col + (((kk & 15) >= 8) ? 16 : 0);
                        int h  = ((kk >= 16) ? 8 : 0) + (kk & 7);
                        ((_Float16*)out1)[(((size_t)(bh * 32 + keyblk) * 2 + ctile) * 32 + lp) * 16 + h] = (_Float16)val;
                    }
                }
            }
        }
    }
}

// -------------------------------------------------------------------------
// Spatial-reduction attention: per (b,head) group: Q[4096,32], K/V[1024,32].
// One wave owns 16 q-rows. QK^T 16x16 tile = ONE v_wmma_f32_16x16x32_f16.
// Two-pass softmax; P re-staged through LDS into A-fragment form; P*V WMMAs.
// K tiles block-cached in LDS (32KB) via GLOBAL_LOAD_ASYNC_TO_LDS_B128.
// grid.x = 32 bh * 32 row-groups; block = 256 threads (8 waves x 16 rows).
// -------------------------------------------------------------------------
__global__ __launch_bounds__(256) void attn_img_k(
    const _Float16* __restrict__ qA, const _Float16* __restrict__ kB,
    const _Float16* __restrict__ vB, float* __restrict__ sum)
{
    __shared__ __align__(32) _Float16 kLds[16384];   // 32 KB: half of K tiles
    __shared__ __align__(32) _Float16 pSh[8 * 512];  // 8 KB:  per-wave P staging
    __shared__ __align__(32) float    oSh[8 * 512];  // 16 KB: per-wave O transpose

    const int tid  = threadIdx.x;
    const int lane = tid & 31;
    const int w    = tid >> 5;
    const int bh     = blockIdx.x >> 5;
    const int rowblk = (blockIdx.x & 31) * 8 + w;
    const float SCALE = 0.17677669529663687f;  // 32^-0.5

    const v16h aq = *(const v16h*)(qA + ((size_t)(bh * 256 + rowblk) * 32 + lane) * 16);

    // ---- pass 1: row maxima ----
    float vmax[8];
    #pragma unroll
    for (int r = 0; r < 8; ++r) vmax[r] = -3.0e38f;
    v8f zero = {};
    for (int kh = 0; kh < 2; ++kh) {
        __syncthreads();
        stage_lds_async_32k(kLds, kB + (size_t)bh * 32768 + (size_t)kh * 16384, tid);
        __syncthreads();
        for (int kt = 0; kt < 32; ++kt) {
            v16h kb = *(const v16h*)(&kLds[(kt * 32 + lane) * 16]);
            v8f s = __builtin_amdgcn_wmma_f32_16x16x32_f16(
                false, aq, false, kb, (short)0, zero, false, false);
            #pragma unroll
            for (int r = 0; r < 8; ++r) vmax[r] = fmaxf(vmax[r], s[r]);
        }
    }
    #pragma unroll
    for (int r = 0; r < 8; ++r) {   // reduce within 16-lane halves (row striping)
        float v = vmax[r];
        v = fmaxf(v, __shfl_xor(v, 1, 32));
        v = fmaxf(v, __shfl_xor(v, 2, 32));
        v = fmaxf(v, __shfl_xor(v, 4, 32));
        v = fmaxf(v, __shfl_xor(v, 8, 32));
        vmax[r] = v;
    }

    // ---- pass 2: exp, row sums, P*V ----
    float vsum[8];
    #pragma unroll
    for (int r = 0; r < 8; ++r) vsum[r] = 0.0f;
    v8f o0 = zero, o1 = zero;

    for (int kh = 0; kh < 2; ++kh) {
        __syncthreads();
        stage_lds_async_32k(kLds, kB + (size_t)bh * 32768 + (size_t)kh * 16384, tid);
        __syncthreads();
        for (int ktp = 0; ktp < 16; ++ktp) {
            #pragma unroll
            for (int sub = 0; sub < 2; ++sub) {
                int kt = ktp * 2 + sub;
                v16h kb = *(const v16h*)(&kLds[(kt * 32 + lane) * 16]);
                v8f s = __builtin_amdgcn_wmma_f32_16x16x32_f16(
                    false, aq, false, kb, (short)0, zero, false, false);
                #pragma unroll
                for (int r = 0; r < 8; ++r) {
                    float p = __expf((s[r] - vmax[r]) * SCALE);
                    vsum[r] += p;
                    int m  = r + ((lane >= 16) ? 8 : 0);
                    int lp = m + (((lane & 15) >= 8) ? 16 : 0);
                    int h  = sub * 8 + (lane & 7);
                    pSh[w * 512 + lp * 16 + h] = (_Float16)p;   // A-frag layout
                }
            }
            __builtin_amdgcn_wave_barrier();
            v16h pa = *(const v16h*)(&pSh[w * 512 + lane * 16]);
            int ktg = kh * 16 + ktp;
            const _Float16* vbase = vB + ((size_t)(bh * 32 + ktg) * 2) * 512;
            // prefetch next V block (global_prefetch_b8)
            __builtin_prefetch(vbase + 1024, 0, 0);
            v16h vb0 = *(const v16h*)(vbase + lane * 16);
            v16h vb1 = *(const v16h*)(vbase + 512 + lane * 16);
            o0 = __builtin_amdgcn_wmma_f32_16x16x32_f16(false, pa, false, vb0, (short)0, o0, false, false);
            o1 = __builtin_amdgcn_wmma_f32_16x16x32_f16(false, pa, false, vb1, (short)0, o1, false, false);
        }
    }

    #pragma unroll
    for (int r = 0; r < 8; ++r) {
        float v = vsum[r];
        v += __shfl_xor(v, 1, 32);
        v += __shfl_xor(v, 2, 32);
        v += __shfl_xor(v, 4, 32);
        v += __shfl_xor(v, 8, 32);
        vsum[r] = 1.0f / v;
    }
    // transpose O through LDS for coalesced [b][d][pos] stores
    #pragma unroll
    for (int r = 0; r < 8; ++r) {
        int m  = r + ((lane >= 16) ? 8 : 0);
        int c0 = lane & 15;
        oSh[w * 512 + c0 * 16 + m]        = o0[r] * vsum[r];
        oSh[w * 512 + (16 + c0) * 16 + m] = o1[r] * vsum[r];
    }
    __builtin_amdgcn_wave_barrier();
    int b = bh >> 3, head = bh & 7;
    float4* dstg = (float4*)(sum + (size_t)(b * 256 + head * 32 + lane) * 4096 + rowblk * 16);
    const float4* srcl = (const float4*)(&oSh[w * 512 + lane * 16]);
    #pragma unroll
    for (int i = 0; i < 4; ++i) dstg[i] = srcl[i];
}

// -------------------------------------------------------------------------
// Batch-branch attention: per (xy, head) slice, 4x4 attention over batch.
// Far below WMMA granularity -> scalar VALU. Accumulates into sum buffer.
// -------------------------------------------------------------------------
__global__ __launch_bounds__(256) void attn_batch_k(
    const float* __restrict__ qb, const float* __restrict__ kvb,
    float* __restrict__ sum)
{
    const int t    = blockIdx.x * 256 + threadIdx.x;  // 0..32767
    const int xy   = t & 4095;
    const int head = t >> 12;                          // 0..7
    const float* kb = kvb;
    const float* vb = kvb + (size_t)256 * 16384;
    const float SCALE = 0.17677669529663687f;

    float sc[4][4];
    #pragma unroll
    for (int bq = 0; bq < 4; ++bq)
        #pragma unroll
        for (int bk = 0; bk < 4; ++bk) sc[bq][bk] = 0.0f;

    for (int c = 0; c < 32; ++c) {
        size_t row = (size_t)(head * 32 + c) * 16384;
        float qv[4], kv[4];
        #pragma unroll
        for (int b = 0; b < 4; ++b) {
            qv[b] = qb[row + b * 4096 + xy];
            kv[b] = kb[row + b * 4096 + xy];
        }
        #pragma unroll
        for (int bq = 0; bq < 4; ++bq)
            #pragma unroll
            for (int bk = 0; bk < 4; ++bk) sc[bq][bk] += qv[bq] * kv[bk];
    }
    #pragma unroll
    for (int bq = 0; bq < 4; ++bq) {
        float m = sc[bq][0];
        #pragma unroll
        for (int bk = 1; bk < 4; ++bk) m = fmaxf(m, sc[bq][bk]);
        float s = 0.0f;
        #pragma unroll
        for (int bk = 0; bk < 4; ++bk) {
            sc[bq][bk] = __expf((sc[bq][bk] - m) * SCALE);
            s += sc[bq][bk];
        }
        float inv = 1.0f / s;
        #pragma unroll
        for (int bk = 0; bk < 4; ++bk) sc[bq][bk] *= inv;
    }
    for (int c = 0; c < 32; ++c) {
        size_t row = (size_t)(head * 32 + c) * 16384;
        float vv[4];
        #pragma unroll
        for (int b = 0; b < 4; ++b) vv[b] = vb[row + b * 4096 + xy];
        #pragma unroll
        for (int bq = 0; bq < 4; ++bq) {
            float o = sc[bq][0] * vv[0] + sc[bq][1] * vv[1] +
                      sc[bq][2] * vv[2] + sc[bq][3] * vv[3];
            size_t di = (size_t)(bq * 256 + head * 32 + c) * 4096 + xy;
            sum[di] += o;   // fuse out_img + out_batch (runs after attn_img_k)
        }
    }
}

// -------------------------------------------------------------------------
extern "C" void kernel_launch(void* const* d_in, const int* in_sizes, int n_in,
                              void* d_out, int out_size, void* d_ws, size_t ws_size,
                              hipStream_t stream) {
    const float* x    = (const float*)d_in[0];
    const float* Wq   = (const float*)d_in[1];
    const float* Wkv  = (const float*)d_in[2];
    const float* Wqb  = (const float*)d_in[3];
    const float* Wkvb = (const float*)d_in[4];
    const float* Wout = (const float*)d_in[5];
    float* out = (float*)d_out;

    char* ws = (char*)d_ws;
    _Float16* qA  = (_Float16*)(ws);                       //  8 MB: Q, A-frag layout
    _Float16* kB  = (_Float16*)(ws + ((size_t)8  << 20));  //  2 MB: K, B-frag layout
    _Float16* vB  = (_Float16*)(ws + ((size_t)10 << 20));  //  2 MB: V, B-frag layout
    float*    qb  = (float*)   (ws + ((size_t)12 << 20));  // 16 MB
    float*    kvb = (float*)   (ws + ((size_t)28 << 20));  // 32 MB
    float*    sum = (float*)   (ws + ((size_t)60 << 20));  // 16 MB (img + batch)

    dim3 blk(256);
    // q = Wq @ x  -> qA (f16, attention A-fragment layout)
    gemm_wmma_k<0, 1><<<dim3(128, 2), blk, 0, stream>>>(Wq, x, qA, nullptr, 256, 16384);
    // kv = stride-2 2x2 conv -> kB / vB (f16, B-fragment layouts)
    gemm_wmma_k<1, 2><<<dim3(32, 4), blk, 0, stream>>>(Wkv, x, kB, vB, 1024, 4096);
    // qb = Wqb @ x, kvb = Wkvb @ x (f32 plain layout)
    gemm_wmma_k<0, 0><<<dim3(128, 2), blk, 0, stream>>>(Wqb, x, qb, nullptr, 256, 16384);
    gemm_wmma_k<0, 0><<<dim3(128, 4), blk, 0, stream>>>(Wkvb, x, kvb, nullptr, 256, 16384);
    // spatial-reduction attention -> sum (writes every element)
    attn_img_k<<<dim3(1024), blk, 0, stream>>>(qA, kB, vB, sum);
    // batch-branch attention, accumulates into sum
    attn_batch_k<<<dim3(128), blk, 0, stream>>>(qb, kvb, sum);
    // out = Wout @ (out_img + out_batch)
    gemm_wmma_k<2, 0><<<dim3(128, 2), blk, 0, stream>>>(Wout, sum, out, nullptr, 256, 16384);
}
